// BigraphGATv2Model_51831665328519
// MI455X (gfx1250) — matrix operand order repaired and checked by reading, not compiled, not run on hardware
//
#include <hip/hip_runtime.h>
#include <hip/hip_bf16.h>

typedef __attribute__((ext_vector_type(16))) _Float16 v16h;
typedef __attribute__((ext_vector_type(8)))  _Float16 v8h;
typedef __attribute__((ext_vector_type(8)))  float    v8f;

#define D 128
#define NEG_SLOPE 0.2f

__device__ __forceinline__ int get_nn(const int* __restrict__ p, int useNItem, int N) {
  return useNItem ? p[0] : N;
}

// ---------------------------------------------------------------- utility
__global__ void k_copy_f32(const float* __restrict__ src, float* __restrict__ dst, int n) {
  int i = blockIdx.x * blockDim.x + threadIdx.x;
  if (i < n) dst[i] = src[i];
}

// h (f32) -> h16 (f16), zero padding rows >= Nn
__global__ void k_conv_h16(const float* __restrict__ hIn, _Float16* __restrict__ h16,
                           const int* __restrict__ nitemp, int useNItem, int N, int Npad) {
  int idx = blockIdx.x * blockDim.x + threadIdx.x;
  if (idx >= Npad * D) return;
  int i = idx >> 7;
  int Nn = get_nn(nitemp, useNItem, N);
  h16[idx] = (i < Nn) ? (_Float16)hIn[idx] : (_Float16)0.0f;
}

// Repack W[128][128] (row major, h@W) into WMMA B-fragment order:
// frag f = kb*8+nt, per-lane 16 halfs: K=kb*32+(lane>=16?8:0)+(j&7)+(j>=8?16:0), Ncol=nt*16+(lane&15)
__global__ void k_repack_w(const float* __restrict__ W, _Float16* __restrict__ rep) {
  int idx = blockIdx.x * blockDim.x + threadIdx.x;  // 32 frags * 512 halfs
  if (idx >= 32 * 512) return;
  int f = idx >> 9, rrem = idx & 511;
  int lane = rrem >> 4, j = rrem & 15;
  int kb = f >> 3, nt = f & 7;
  int K = kb * 32 + ((lane & 16) ? 8 : 0) + (j & 7) + ((j & 8) ? 16 : 0);
  int n = nt * 16 + (lane & 15);
  rep[idx] = (_Float16)W[K * D + n];
}

// ---------------------------------------------------------------- GEMM: xl = h@Wl + bl ; xr = h@Wr + br
// 256 threads = 8 waves; each wave computes one 16-row tile x 128 cols for BOTH outputs.
__global__ __launch_bounds__(256) void k_gemm_xlxr(
    const _Float16* __restrict__ h16,
    const _Float16* __restrict__ wlRep, const _Float16* __restrict__ wrRep,
    const float* __restrict__ bl, const float* __restrict__ br,
    float* __restrict__ xl, float* __restrict__ xr,
    const int* __restrict__ nitemp, int useNItem, int N) {
  __shared__ _Float16 sW[2][32 * 512];  // 2 x 32KB repacked weights

  const int tid = threadIdx.x;
  // cooperative LDS fill (8192 dwords per matrix)
  {
    const uint32_t* gl = (const uint32_t*)wlRep;
    const uint32_t* gr = (const uint32_t*)wrRep;
    uint32_t* sl = (uint32_t*)&sW[0][0];
    uint32_t* sr = (uint32_t*)&sW[1][0];
    for (int i = tid; i < 8192; i += 256) { sl[i] = gl[i]; sr[i] = gr[i]; }
  }
  __syncthreads();

  const int Nn = get_nn(nitemp, useNItem, N);
  const int wave = tid >> 5, lane = tid & 31;
  const int mtile = blockIdx.x * 8 + wave;
  const int rowBase = mtile * 16;
  if (rowBase >= Nn) return;  // wave-uniform exit; EXEC stays all-1 for WMMA

  v8f zero = {};
  v8f accL[8], accR[8];
#pragma unroll
  for (int t = 0; t < 8; ++t) { accL[t] = zero; accR[t] = zero; }

  // A fragment addressing (16-bit A 16x32 layout):
  // lane row = rowBase + (lane&15); chunk base K = kb*32 + ((lane&16)>>1)*? -> (lane>=16 ? 8 : 0)
  const int arow = rowBase + (lane & 15);
  const _Float16* aptr = h16 + (size_t)arow * D + ((lane & 16) >> 1);

#pragma unroll
  for (int kb = 0; kb < 4; ++kb) {
    union { v16h v; v8h h[2]; } a;
    a.h[0] = *(const v8h*)(aptr + kb * 32);        // K = c0 .. c0+7
    a.h[1] = *(const v8h*)(aptr + kb * 32 + 16);   // K = c0+16 .. c0+23
#pragma unroll
    for (int nt = 0; nt < 8; ++nt) {
      const int f = kb * 8 + nt;
      v16h bL = *(const v16h*)(&sW[0][f * 512 + lane * 16]);
      v16h bR = *(const v16h*)(&sW[1][f * 512 + lane * 16]);
      accL[nt] = __builtin_amdgcn_wmma_f32_16x16x32_f16(false, a.v, false, bL,
                                                        (short)0, accL[nt], false, false);
      accR[nt] = __builtin_amdgcn_wmma_f32_16x16x32_f16(false, a.v, false, bR,
                                                        (short)0, accR[nt], false, false);
    }
  }

  // C/D layout: VGPR r, lane<16 -> (M=r, N=lane); lane>=16 -> (M=r+8, N=lane-16)
  const int colB = lane & 15;
  const int rowOff = (lane >= 16) ? 8 : 0;
#pragma unroll
  for (int nt = 0; nt < 8; ++nt) {
    const int col = nt * 16 + colB;
    const float bLv = bl[col], bRv = br[col];
#pragma unroll
    for (int rr = 0; rr < 8; ++rr) {
      const int row = rowBase + rr + rowOff;
      if (row < Nn) {
        xl[(size_t)row * D + col] = accL[nt][rr] + bLv;
        xr[(size_t)row * D + col] = accR[nt][rr] + bRv;
      }
    }
  }
}

// ---------------------------------------------------------------- per-layer init
// out rows [0,Nn) = bias ; rows [Nn,N) = carry-over from hIn (partial layers); reset m/s.
__global__ void k_init_layer(float* __restrict__ outBuf, const float* __restrict__ hIn,
                             const float* __restrict__ bias, uint32_t* __restrict__ mkey,
                             float* __restrict__ ssum, const int* __restrict__ nitemp,
                             int useNItem, int N) {
  int idx = blockIdx.x * blockDim.x + threadIdx.x;
  if (idx >= N * D) return;
  int i = idx >> 7, c = idx & 127;
  int Nn = get_nn(nitemp, useNItem, N);
  outBuf[idx] = (i < Nn) ? bias[c] : hIn[idx];
  if (c == 0) { mkey[i] = 0u; ssum[i] = 0.0f; }
}

__device__ __forceinline__ uint32_t f32_key(float f) {
  uint32_t u = __float_as_uint(f);
  return (u & 0x80000000u) ? ~u : (u | 0x80000000u);
}
__device__ __forceinline__ float key_f32(uint32_t k) {
  uint32_t u = (k & 0x80000000u) ? (k ^ 0x80000000u) : ~k;
  return __uint_as_float(u);
}

// ---------------------------------------------------------------- edge pass 1: scores + segment max
// one wave32 per edge; lane handles float4 of the 128-wide row
__global__ void k_edge_score(const float* __restrict__ xl, const float* __restrict__ xr,
                             const int* __restrict__ srcA, const int* __restrict__ dstA,
                             const float* __restrict__ eattr, const float* __restrict__ We,
                             const float* __restrict__ att, float* __restrict__ escore,
                             uint32_t* __restrict__ mkey, const int* __restrict__ nitemp,
                             int useNItem, int N, int E) {
  int gid = blockIdx.x * blockDim.x + threadIdx.x;
  int e = gid >> 5, lane = gid & 31;
  int Nn = get_nn(nitemp, useNItem, N);
  if (e >= E + Nn) return;
  int s, d; float ea;
  if (e < E) { s = srcA[e]; d = dstA[e]; ea = eattr[e]; }
  else       { s = d = e - E; ea = 1.0f; }

  const float4 xl4 = *(const float4*)(xl + (size_t)s * D + lane * 4);
  const float4 xr4 = *(const float4*)(xr + (size_t)d * D + lane * 4);
  const float4 we4 = *(const float4*)(We + lane * 4);
  const float4 at4 = *(const float4*)(att + lane * 4);

  float p = 0.0f, z;
  z = xl4.x + xr4.x + ea * we4.x; z = (z > 0.0f) ? z : NEG_SLOPE * z; p += z * at4.x;
  z = xl4.y + xr4.y + ea * we4.y; z = (z > 0.0f) ? z : NEG_SLOPE * z; p += z * at4.y;
  z = xl4.z + xr4.z + ea * we4.z; z = (z > 0.0f) ? z : NEG_SLOPE * z; p += z * at4.z;
  z = xl4.w + xr4.w + ea * we4.w; z = (z > 0.0f) ? z : NEG_SLOPE * z; p += z * at4.w;
#pragma unroll
  for (int off = 16; off > 0; off >>= 1) p += __shfl_xor(p, off, 32);

  if (lane == 0) {
    escore[e] = p;
    atomicMax(&mkey[d], f32_key(p));
  }
}

// ---------------------------------------------------------------- edge pass 2: exp + segment sum
__global__ void k_edge_exp(const float* __restrict__ escore, float* __restrict__ aexp,
                           float* __restrict__ ssum, const uint32_t* __restrict__ mkey,
                           const int* __restrict__ dstA, const int* __restrict__ nitemp,
                           int useNItem, int N, int E) {
  int e = blockIdx.x * blockDim.x + threadIdx.x;
  int Nn = get_nn(nitemp, useNItem, N);
  if (e >= E + Nn) return;
  int d = (e < E) ? dstA[e] : (e - E);
  float a = __expf(escore[e] - key_f32(mkey[d]));
  aexp[e] = a;
  atomicAdd(&ssum[d], a);
}

// ---------------------------------------------------------------- edge pass 3: weighted scatter-add
__global__ void k_edge_agg(const float* __restrict__ xl, const float* __restrict__ aexp,
                           const float* __restrict__ ssum, const int* __restrict__ srcA,
                           const int* __restrict__ dstA, float* __restrict__ outBuf,
                           const int* __restrict__ nitemp, int useNItem, int N, int E) {
  int gid = blockIdx.x * blockDim.x + threadIdx.x;
  int e = gid >> 5, lane = gid & 31;
  int Nn = get_nn(nitemp, useNItem, N);
  if (e >= E + Nn) return;
  int s, d;
  if (e < E) { s = srcA[e]; d = dstA[e]; }
  else       { s = d = e - E; }
  float alpha = aexp[e] / ssum[d];
  const float4 v = *(const float4*)(xl + (size_t)s * D + lane * 4);
  float* o = outBuf + (size_t)d * D + lane * 4;
  atomicAdd(o + 0, alpha * v.x);
  atomicAdd(o + 1, alpha * v.y);
  atomicAdd(o + 2, alpha * v.z);
  atomicAdd(o + 3, alpha * v.w);
}

// ================================================================ host
extern "C" void kernel_launch(void* const* d_in, const int* in_sizes, int n_in,
                              void* d_out, int out_size, void* d_ws, size_t ws_size,
                              hipStream_t stream) {
  (void)n_in; (void)out_size; (void)ws_size;
  const float* x      = (const float*)d_in[0];
  const int*   ei_ii  = (const int*)d_in[1];
  const float* ea_ii  = (const float*)d_in[2];
  const int*   ei_uiu = (const int*)d_in[3];
  const float* ea_uiu = (const float*)d_in[4];
  const int*   nitem  = (const int*)d_in[5];
  const float* Wl     = (const float*)d_in[6];
  const float* blp    = (const float*)d_in[7];
  const float* Wr     = (const float*)d_in[8];
  const float* brp    = (const float*)d_in[9];
  const float* Wep    = (const float*)d_in[10];
  const float* attp   = (const float*)d_in[11];
  const float* biasp  = (const float*)d_in[12];

  const int N    = in_sizes[0] / D;
  const int Eii  = in_sizes[2];
  const int Euiu = in_sizes[4];
  const int Npad = (N + 15) & ~15;
  const int Emax = (Eii > Euiu ? Eii : Euiu);

  // workspace carve-up (256B aligned)
  char* w = (char*)d_ws; size_t off = 0;
  auto take = [&](size_t bytes) -> void* {
    void* p = w + off; off = (off + bytes + 255) & ~(size_t)255; return p;
  };
  float*     hA     = (float*)take((size_t)Npad * D * 4);
  float*     hB     = (float*)take((size_t)Npad * D * 4);
  float*     xlB    = (float*)take((size_t)Npad * D * 4);
  float*     xrB    = (float*)take((size_t)Npad * D * 4);
  _Float16*  h16    = (_Float16*)take((size_t)Npad * D * 2);
  _Float16*  wlRep  = (_Float16*)take(32 * 512 * 2);
  _Float16*  wrRep  = (_Float16*)take(32 * 512 * 2);
  uint32_t*  mkey   = (uint32_t*)take((size_t)Npad * 4);
  float*     ssum   = (float*)take((size_t)Npad * 4);
  float*     escore = (float*)take((size_t)(Emax + Npad) * 4);
  float*     aexp   = (float*)take((size_t)(Emax + Npad) * 4);

  auto cdiv = [](long a, long b) -> int { return (int)((a + b - 1) / b); };

  // h0 = x
  k_copy_f32<<<cdiv((long)N * D, 256), 256, 0, stream>>>(x, hA, N * D);

  float* hs_in [4] = { hA, hB, hA, hB };
  float* hs_out[4] = { hB, hA, hB, (float*)d_out };

  for (int l = 0; l < 4; ++l) {
    const int partial = (l < 2) ? 1 : 0;
    const int E       = partial ? Eii : Euiu;
    const int* srcp   = partial ? ei_ii : ei_uiu;
    const int* dstp   = srcp + E;
    const float* eap  = partial ? ea_ii : ea_uiu;
    float* hIn  = hs_in[l];
    float* hOut = hs_out[l];
    const long totMax = (long)E + N;

    k_conv_h16<<<cdiv((long)Npad * D, 256), 256, 0, stream>>>(hIn, h16, nitem, partial, N, Npad);
    k_repack_w<<<64, 256, 0, stream>>>(Wl + (size_t)l * D * D, wlRep);
    k_repack_w<<<64, 256, 0, stream>>>(Wr + (size_t)l * D * D, wrRep);
    k_gemm_xlxr<<<cdiv(Npad / 16, 8), 256, 0, stream>>>(
        h16, wlRep, wrRep, blp + l * D, brp + l * D, xlB, xrB, nitem, partial, N);
    k_init_layer<<<cdiv((long)N * D, 256), 256, 0, stream>>>(
        hOut, hIn, biasp + l * D, mkey, ssum, nitem, partial, N);
    k_edge_score<<<cdiv(totMax * 32, 256), 256, 0, stream>>>(
        xlB, xrB, srcp, dstp, eap, Wep + l * D, attp + l * D, escore, mkey, nitem, partial, N, E);
    k_edge_exp<<<cdiv(totMax, 256), 256, 0, stream>>>(
        escore, aexp, ssum, mkey, dstp, nitem, partial, N, E);
    k_edge_agg<<<cdiv(totMax * 32, 256), 256, 0, stream>>>(
        xlB, aexp, ssum, srcp, dstp, hOut, nitem, partial, N, E);
  }
}
